// ActorNet_42219528520046
// MI455X (gfx1250) — matrix operand order repaired and checked by reading, not compiled, not run on hardware
//
#include <hip/hip_runtime.h>
#include <hip/hip_bf16.h>

#define NN 50000
#define EE 200000
#define TT 2000

typedef _Float16 v16h __attribute__((ext_vector_type(16)));
typedef _Float16 v8h  __attribute__((ext_vector_type(8)));
typedef float    v8f  __attribute__((ext_vector_type(8)));

__device__ __forceinline__ float sigm(float x) { return 1.0f / (1.0f + __expf(-x)); }

// ---------------------------------------------------------------- init / prep

__global__ void k_init(float* agg, unsigned* cnt, float* state) {
    size_t i = (size_t)blockIdx.x * 256 + threadIdx.x;
    if (i < (size_t)NN * 32) agg[i] = 0.0f;
    if (i < NN)  cnt[i] = 0u;
    if (i < 256) state[i] = 0.0f;
}

__global__ void k_lin0(const float* __restrict__ x, const float* __restrict__ W0,
                       const float* __restrict__ b0, float* __restrict__ out,
                       _Float16* __restrict__ outh) {
    size_t i = (size_t)blockIdx.x * 256 + threadIdx.x;
    if (i >= (size_t)NN * 32) return;
    int n = (int)(i >> 5), f = (int)(i & 31);
    float v = b0[f] + x[n * 3 + 0] * W0[f] + x[n * 3 + 1] * W0[32 + f] + x[n * 3 + 2] * W0[64 + f];
    v = fmaxf(v, 0.0f);
    out[i]  = v;
    outh[i] = (_Float16)v;
}

__global__ void k_edge(const float* __restrict__ ea, const float* __restrict__ W1,
                       const float* __restrict__ b1, _Float16* __restrict__ heh) {
    size_t i = (size_t)blockIdx.x * 256 + threadIdx.x;
    if (i >= (size_t)EE * 32) return;
    int e = (int)(i >> 5), f = (int)(i & 31);
    float s = b1[f];
    #pragma unroll
    for (int k = 0; k < 7; k++) s += ea[e * 7 + k] * W1[k * 32 + f];
    heh[i] = (_Float16)fmaxf(s, 0.0f);
}

// Swizzle nn2_W (32x1024 f32) and nn2_b (1024) into CDNA5 16x16x32 B-fragment order (f16).
__global__ void k_cvt(const float* __restrict__ nn2W, const float* __restrict__ nn2b,
                      _Float16* __restrict__ wfrag, _Float16* __restrict__ bfrag) {
    int i = blockIdx.x * 256 + threadIdx.x;
    if (i < 16384) {                       // weight fragments: 32 ksteps x 2 ntiles x 32 lanes x 8 dw
        int j  = i & 7;
        int l  = (i >> 3) & 31;
        int tt = (i >> 8) & 1;
        int kk = i >> 9;
        int d0 = 2 * j + ((l < 16) ? 0 : 8) + ((j < 4) ? 0 : 8);
        int f  = tt * 16 + (l & 15);
        wfrag[2 * i]     = (_Float16)nn2W[kk * 1024 + d0 * 32 + f];
        wfrag[2 * i + 1] = (_Float16)nn2W[kk * 1024 + (d0 + 1) * 32 + f];
    } else if (i < 16896) {                // bias fragments: 2 ntiles x 32 lanes x 8 dw
        int b  = i - 16384;
        int j  = b & 7;
        int l  = (b >> 3) & 31;
        int tt = (b >> 8) & 1;
        int d0 = 2 * j + ((l < 16) ? 0 : 8) + ((j < 4) ? 0 : 8);
        int f  = tt * 16 + (l & 15);
        bfrag[2 * b]     = (_Float16)nn2b[d0 * 32 + f];
        bfrag[2 * b + 1] = (_Float16)nn2b[(d0 + 1) * 32 + f];
    }
}

__global__ void k_cnt(const int* __restrict__ dst, unsigned* __restrict__ cnt) {
    int i = blockIdx.x * 256 + threadIdx.x;
    if (i < EE) atomicAdd(&cnt[dst[i]], 1u);
}

__global__ void k_rden(const unsigned* __restrict__ cnt, float* __restrict__ rden) {
    int i = blockIdx.x * 256 + threadIdx.x;
    if (i < NN) { unsigned c = cnt[i]; rden[i] = 1.0f / (float)(c ? c : 1u); }
}

// ------------------------------------------------ WMMA message passing kernel
// msg[e,f] = sum_{k,d} he[e,k] * out[src_e,d] * nn2W[k, d*32+f]  (+ bias via he==1)
// 16-edge tiles; A fragment built in-register as (out_src pattern) * he[k] scalar.

__global__ void __launch_bounds__(256) k_msg(
    const _Float16* __restrict__ heh, const _Float16* __restrict__ outh,
    const v16h* __restrict__ wfrag, const v16h* __restrict__ bfrag,
    const int* __restrict__ src, const int* __restrict__ dst,
    float* __restrict__ agg) {
    __shared__ v16h B[2048];                       // 64 KB: 32 ksteps x 2 ntiles x 32 lanes
    int t = threadIdx.x;
    for (int i = t; i < 2048; i += 256) B[i] = wfrag[i];
    __syncthreads();

    int wid = blockIdx.x * 8 + (t >> 5);           // wave-uniform tile id
    if (wid >= EE / 16) return;
    int lane = t & 31;
    int erow = wid * 16 + (lane & 15);
    int s    = src[erow];
    int off  = (lane < 16) ? 0 : 8;

    // A-pattern per ISA 16-bit A 16x32 layout: halves d = off..off+7, 16+off..16+off+7
    v8h a_lo = *(const v8h*)(outh + (size_t)s * 32 + off);
    v8h a_hi = *(const v8h*)(outh + (size_t)s * 32 + 16 + off);
    v16h ap;
    #pragma unroll
    for (int i = 0; i < 8; i++) { ap[i] = a_lo[i]; ap[8 + i] = a_hi[i]; }

    v16h h0 = *(const v16h*)(heh + (size_t)erow * 32);
    v16h h1 = *(const v16h*)(heh + (size_t)erow * 32 + 16);

    v8f c0 = {}; v8f c1 = {};
    #pragma unroll
    for (int kk = 0; kk < 32; kk++) {
        _Float16 hv = (kk < 16) ? h0[kk] : h1[kk - 16];
        v16h a  = ap * hv;                         // v_pk_mul_f16 x8
        v16h b0 = B[(kk * 2 + 0) * 32 + lane];
        c0 = __builtin_amdgcn_wmma_f32_16x16x32_f16(false, a, false, b0, (short)0, c0, false, false);
        v16h b1 = B[(kk * 2 + 1) * 32 + lane];
        c1 = __builtin_amdgcn_wmma_f32_16x16x32_f16(false, a, false, b1, (short)0, c1, false, false);
    }
    {   // nn2_b contribution: extra K-step with he == 1
        v16h b0 = bfrag[lane];
        c0 = __builtin_amdgcn_wmma_f32_16x16x32_f16(false, ap, false, b0, (short)0, c0, false, false);
        v16h b1 = bfrag[32 + lane];
        c1 = __builtin_amdgcn_wmma_f32_16x16x32_f16(false, ap, false, b1, (short)0, c1, false, false);
    }
    // scatter C fragments: VGPR i -> edge row i (lanes<16) / 8+i (lanes>=16), col = lane%16
    int mbase = wid * 16 + ((lane < 16) ? 0 : 8);
    int fc    = lane & 15;
    #pragma unroll
    for (int i = 0; i < 8; i++) {
        int d = dst[mbase + i];
        atomicAdd(agg + (size_t)d * 32 + fc,      c0[i]);
        atomicAdd(agg + (size_t)d * 32 + 16 + fc, c1[i]);
    }
}

// ----------------------------------------------------- fused node update + GRU

__global__ void __launch_bounds__(256) k_node(
    float* __restrict__ out, _Float16* __restrict__ outh, float* __restrict__ agg,
    const float* __restrict__ rden, const float* __restrict__ rootW,
    const float* __restrict__ convb, const float* __restrict__ Wih,
    const float* __restrict__ Whh, const float* __restrict__ bih,
    const float* __restrict__ bhh) {
    __shared__ float Wr[1024];
    __shared__ float WiT[3072];        // transposed [d][g] for conflict-free reads
    __shared__ float WhT[3072];
    __shared__ float osh[8][33];
    __shared__ float msh[8][33];
    int t = threadIdx.x;
    for (int i = t; i < 1024; i += 256) Wr[i] = rootW[i];
    for (int i = t; i < 3072; i += 256) {
        int g = i / 32, d = i % 32;
        WiT[d * 96 + g] = Wih[i];
        WhT[d * 96 + g] = Whh[i];
    }
    int r = t >> 5, f = t & 31;
    int n = blockIdx.x * 8 + r;
    osh[r][f] = out[(size_t)n * 32 + f];
    __syncthreads();

    float s = agg[(size_t)n * 32 + f] * rden[n] + convb[f];
    for (int d = 0; d < 32; d++) s += osh[r][d] * Wr[d * 32 + f];
    msh[r][f] = fmaxf(s, 0.0f);
    agg[(size_t)n * 32 + f] = 0.0f;    // leave zeroed for next iteration
    __syncthreads();

    float gir = bih[f], giz = bih[32 + f], gin = bih[64 + f];
    float ghr = bhh[f], ghz = bhh[32 + f], ghn = bhh[64 + f];
    for (int d = 0; d < 32; d++) {
        float md = msh[r][d], hd = osh[r][d];
        gir += md * WiT[d * 96 + f];
        giz += md * WiT[d * 96 + 32 + f];
        gin += md * WiT[d * 96 + 64 + f];
        ghr += hd * WhT[d * 96 + f];
        ghz += hd * WhT[d * 96 + 32 + f];
        ghn += hd * WhT[d * 96 + 64 + f];
    }
    float rr = sigm(gir + ghr);
    float zz = sigm(giz + ghz);
    float nn = tanhf(gin + rr * ghn);
    float hn = (1.0f - zz) * nn + zz * osh[r][f];
    out[(size_t)n * 32 + f]  = hn;
    outh[(size_t)n * 32 + f] = (_Float16)hn;
}

// ------------------------------------------------------- Set2Set / LSTM / head
// state layout: [0:64) q_star, [64:96) hs, [96:128) cs, [128:160) r_raw,
//               [160] S, [168:200) hx, [200:232) cx

__global__ void k_lstm(const float* __restrict__ Wih, const float* __restrict__ Whh,
                       const float* __restrict__ bih, const float* __restrict__ bhh,
                       const float* __restrict__ in, int in_dim,
                       float* __restrict__ h, float* __restrict__ c,
                       float* zero_buf, int zero_n, float* out_hc) {
    __shared__ float g[128];
    __shared__ float hsh[32];
    int t = threadIdx.x;               // 128 threads
    if (t < 32) hsh[t] = h[t];
    __syncthreads();
    float s = bih[t] + bhh[t];
    for (int j = 0; j < in_dim; j++) s += in[j] * Wih[t * in_dim + j];
    for (int j = 0; j < 32; j++)     s += hsh[j] * Whh[t * 32 + j];
    g[t] = s;
    __syncthreads();
    if (t < 32) {
        float ig = sigm(g[t]), fg = sigm(g[32 + t]);
        float gg = tanhf(g[64 + t]), og = sigm(g[96 + t]);
        float cn = fg * c[t] + ig * gg;
        float hn = og * tanhf(cn);
        c[t] = cn; h[t] = hn;
        if (out_hc) { out_hc[t] = hn; out_hc[32 + t] = cn; }
    }
    if (zero_buf && t < zero_n) zero_buf[t] = 0.0f;
}

__global__ void __launch_bounds__(256) k_attn(const float* __restrict__ out,
                                              float* __restrict__ state) {
    __shared__ float hsh[32];
    __shared__ float rsh[32];
    __shared__ float Ssh;
    int t = threadIdx.x;
    if (t < 32) { hsh[t] = state[64 + t]; rsh[t] = 0.0f; }
    if (t == 0) Ssh = 0.0f;
    __syncthreads();
    int f = t & 31, slot = t >> 5;
    float pr = 0.0f, pS = 0.0f;
    for (int n = blockIdx.x * 8 + slot; n < NN; n += gridDim.x * 8) {
        const float* row = out + (size_t)n * 32;
        float e = 0.0f;
        for (int d = 0; d < 32; d++) e += row[d] * hsh[d];
        float p = __expf(e);           // logits bounded; no max-shift needed
        pr += p * row[f];
        if (f == 0) pS += p;
    }
    atomicAdd(&rsh[f], pr);
    if (f == 0) atomicAdd(&Ssh, pS);
    __syncthreads();
    if (t < 32) atomicAdd(&state[128 + t], rsh[t]);
    if (t == 0) atomicAdd(&state[160], Ssh);
}

__global__ void k_fin(float* __restrict__ state) {
    int t = threadIdx.x;               // 32 threads
    float S = state[160];
    state[t]      = state[64 + t];
    state[32 + t] = state[128 + t] / S;
}

__global__ void __launch_bounds__(256) k_tors(
    const float* __restrict__ out, const int* __restrict__ nonring,
    const float* __restrict__ state, const float* __restrict__ W1,
    const float* __restrict__ b1, const float* __restrict__ W2,
    const float* __restrict__ b2, float* __restrict__ act) {
    __shared__ float Wsh[160 * 32];
    __shared__ float zsh[8][160];
    __shared__ float zr[8][32];
    int t = threadIdx.x;
    for (int i = t; i < 5120; i += 256) Wsh[i] = W1[i];
    int t0 = blockIdx.x * 8;
    for (int i = t; i < 1280; i += 256) {
        int rr = i / 160, a = i % 160;
        int tt = t0 + rr;
        float v = 0.0f;
        if (tt < TT) {
            if (a < 128) {
                int lin  = a * TT + tt;               // faithful reshape/permute
                int node = nonring[lin >> 5];
                v = out[(size_t)node * 32 + (lin & 31)];
            } else {
                v = state[168 + (a - 128)];           // lstm_out (hx)
            }
        }
        zsh[rr][a] = v;
    }
    __syncthreads();
    {
        int rr = t >> 5, f = t & 31;
        float s = b1[f];
        for (int a = 0; a < 160; a++) s += zsh[rr][a] * Wsh[a * 32 + f];
        zr[rr][f] = fmaxf(s, 0.0f);
    }
    __syncthreads();
    if (t < 96) {
        int rr = t / 12, j = t % 12;
        int tt = t0 + rr;
        float s = b2[j];
        for (int f2 = 0; f2 < 32; f2++) s += zr[rr][f2] * W2[f2 * 12 + j];
        if (tt < TT) act[tt * 12 + j] = s;
    }
}

// --------------------------------------------------------------------- launch

extern "C" void kernel_launch(void* const* d_in, const int* in_sizes, int n_in,
                              void* d_out, int out_size, void* d_ws, size_t ws_size,
                              hipStream_t stream) {
    const float* x       = (const float*)d_in[0];
    const int*   eidx    = (const int*)d_in[1];
    const float* ea      = (const float*)d_in[2];
    const int*   nonring = (const int*)d_in[3];
    const float* lin0W = (const float*)d_in[4];
    const float* lin0b = (const float*)d_in[5];
    const float* nn1W  = (const float*)d_in[6];
    const float* nn1b  = (const float*)d_in[7];
    const float* nn2W  = (const float*)d_in[8];
    const float* nn2b  = (const float*)d_in[9];
    const float* rootW = (const float*)d_in[10];
    const float* convb = (const float*)d_in[11];
    const float* gWih  = (const float*)d_in[12];
    const float* gWhh  = (const float*)d_in[13];
    const float* gbih  = (const float*)d_in[14];
    const float* gbhh  = (const float*)d_in[15];
    const float* sWih  = (const float*)d_in[16];
    const float* sWhh  = (const float*)d_in[17];
    const float* sbih  = (const float*)d_in[18];
    const float* sbhh  = (const float*)d_in[19];
    const float* mWih  = (const float*)d_in[20];
    const float* mWhh  = (const float*)d_in[21];
    const float* mbih  = (const float*)d_in[22];
    const float* mbhh  = (const float*)d_in[23];
    const float* l1W   = (const float*)d_in[24];
    const float* l1b   = (const float*)d_in[25];
    const float* l2W   = (const float*)d_in[26];
    const float* l2b   = (const float*)d_in[27];

    const int* src = eidx;
    const int* dst = eidx + EE;
    float* outp = (float*)d_out;

    char* w = (char*)d_ws;
    auto carve = [&](size_t bytes) -> void* {
        void* p = (void*)w;
        w += (bytes + 255) & ~(size_t)255;
        return p;
    };
    float*    out   = (float*)carve((size_t)NN * 32 * 4);
    float*    agg   = (float*)carve((size_t)NN * 32 * 4);
    _Float16* outh  = (_Float16*)carve((size_t)NN * 32 * 2);
    _Float16* heh   = (_Float16*)carve((size_t)EE * 32 * 2);
    _Float16* wfrag = (_Float16*)carve(16384 * 4);
    _Float16* bfrag = (_Float16*)carve(512 * 4);
    unsigned* cnt   = (unsigned*)carve((size_t)NN * 4);
    float*    rden  = (float*)carve((size_t)NN * 4);
    float*    state = (float*)carve(256 * 4);

    k_init<<<6250, 256, 0, stream>>>(agg, cnt, state);
    k_lin0<<<6250, 256, 0, stream>>>(x, lin0W, lin0b, out, outh);
    k_edge<<<25000, 256, 0, stream>>>(ea, nn1W, nn1b, heh);
    k_cvt<<<66, 256, 0, stream>>>(nn2W, nn2b, wfrag, bfrag);
    k_cnt<<<(EE + 255) / 256, 256, 0, stream>>>(dst, cnt);
    k_rden<<<(NN + 255) / 256, 256, 0, stream>>>(cnt, rden);

    for (int it = 0; it < 6; it++) {
        k_msg<<<1563, 256, 0, stream>>>(heh, outh, (const v16h*)wfrag, (const v16h*)bfrag,
                                        src, dst, agg);
        k_node<<<6250, 256, 0, stream>>>(out, outh, agg, rden, rootW, convb,
                                         gWih, gWhh, gbih, gbhh);
    }

    for (int it = 0; it < 6; it++) {
        k_lstm<<<1, 128, 0, stream>>>(sWih, sWhh, sbih, sbhh,
                                      state /*q_star*/, 64,
                                      state + 64 /*hs*/, state + 96 /*cs*/,
                                      state + 128 /*r_raw+S*/, 33, nullptr);
        k_attn<<<512, 256, 0, stream>>>(out, state);
        k_fin<<<1, 32, 0, stream>>>(state);
    }

    // memory LSTM: hx/cx -> d_out[24000..24063]
    k_lstm<<<1, 128, 0, stream>>>(mWih, mWhh, mbih, mbhh,
                                  state /*pool*/, 64,
                                  state + 168 /*hx*/, state + 200 /*cx*/,
                                  nullptr, 0, outp + TT * 12);

    k_tors<<<250, 256, 0, stream>>>(out, nonring, state, l1W, l1b, l2W, l2b, outp);
}
